// SeparateGateIntegrator_12730283066056
// MI455X (gfx1250) — matrix operand order repaired and checked by reading, not compiled, not run on hardware
//
#include <hip/hip_runtime.h>
#include <hip/hip_bf16.h>
#include <stdint.h>

// ---------------------------------------------------------------------------
// SeparateGateIntegrator for MI455X (gfx1250, wave32, WMMA, async-to-LDS)
//
//   f = sigmoid(x@Wf + bf); i = sigmoid(x@Wi + bi); b = (x@Wb + bb)/sqrt(D)
//   h_t = (1-f_t) * h_{t-1} + i_t*b_t    (scan over SEQ)
//
// Pass 1: x fp32 -> bf16 (row-major) ; W fp32 -> bf16 transposed [N][K].
// Pass 2: fused triple GEMM: async global->LDS double-buffered staging,
//         v_wmma_f32_16x16x32_bf16 compute, gate epilogue fused.
// Pass 3: per-channel linear recurrence (in place on the drive buffer).
// ---------------------------------------------------------------------------

#define D_MODEL 1024
#define BATCH   4
#define SEQ     4096
#define M_TOT   (BATCH * SEQ)     // 16384 rows

#define BM   128                  // block tile M
#define BN   128                  // block tile N
#define BK   32                   // K step (one bf16 WMMA K)
#define LDSP 40                   // padded LDS row stride (ushorts), 80B = 5*16B
#define BUFB (512 * LDSP * 2)     // bytes per LDS buffer (40960)

typedef __attribute__((ext_vector_type(16))) __bf16 v16bf;
typedef __attribute__((ext_vector_type(8)))  __bf16 v8bf;
typedef __attribute__((ext_vector_type(2)))  __bf16 v2bf;
typedef __attribute__((ext_vector_type(8)))  float  v8f;
typedef __attribute__((ext_vector_type(2)))  float  v2f;

union Frag { uint4 q[2]; v16bf v; };   // 32 bytes = 16 bf16

__device__ __forceinline__ unsigned int pack2(float lo, float hi) {
  v2f in = { lo, hi };
  union { v2bf v; unsigned int u; } r;
  r.v = __builtin_convertvector(in, v2bf);   // v_cvt_pk_bf16_f32
  return r.u;
}

// 64B row copy global->LDS via async tensor path (4 x 16B, ASYNCcnt).
// INST_OFFSET biases both the LDS and global addresses (ISA 08 §4.4).
__device__ __forceinline__ void async_row64(unsigned lds_off, unsigned gvoff,
                                            const unsigned short* gbase) {
  asm volatile(
      "global_load_async_to_lds_b128 %0, %1, %2 offset:0\n\t"
      "global_load_async_to_lds_b128 %0, %1, %2 offset:16\n\t"
      "global_load_async_to_lds_b128 %0, %1, %2 offset:32\n\t"
      "global_load_async_to_lds_b128 %0, %1, %2 offset:48"
      :: "v"(lds_off), "v"(gvoff), "s"(gbase)
      : "memory");
}
__device__ __forceinline__ void wait_async0() {
  asm volatile("s_wait_asynccnt 0x0" ::: "memory");
}

// ---------------- Pass 1a: x fp32 -> bf16 (same layout) ----------------
extern "C" __global__ __launch_bounds__(256)
void cvt_x_kernel(const float* __restrict__ X, unsigned short* __restrict__ Xb) {
  const size_t i = ((size_t)blockIdx.x * 256 + threadIdx.x) * 8;
  union { float4 f4[2]; v8f v; } in;
  in.f4[0] = *(const float4*)(X + i);
  in.f4[1] = *(const float4*)(X + i + 4);
  union { v8bf v; uint4 q; } out;
  out.v = __builtin_convertvector(in.v, v8bf);
  *(uint4*)(Xb + i) = out.q;
}

// ------- Pass 1b: W fp32 [K][N] -> bf16 transposed [N][K], LDS-tiled -------
extern "C" __global__ __launch_bounds__(256)
void cvt_wt_kernel(const float* __restrict__ Wf, const float* __restrict__ Wi,
                   const float* __restrict__ Wb, unsigned short* __restrict__ Wt) {
  __shared__ float Ls[64][65];
  const float* Wsrc = (blockIdx.y == 0) ? Wf : (blockIdx.y == 1) ? Wi : Wb;
  unsigned short* Wdst = Wt + (size_t)blockIdx.y * D_MODEL * D_MODEL;
  const int kBase = (blockIdx.x & 15) * 64;
  const int nBase = (blockIdx.x >> 4) * 64;
  const int tid = threadIdx.x;
  {
    const int kk = tid >> 2;
    const int nn = (tid & 3) * 16;
    const float* p = Wsrc + (size_t)(kBase + kk) * D_MODEL + nBase + nn;
#pragma unroll
    for (int j = 0; j < 4; ++j) {
      float4 v = *(const float4*)(p + j * 4);
      Ls[kk][nn + j * 4 + 0] = v.x;
      Ls[kk][nn + j * 4 + 1] = v.y;
      Ls[kk][nn + j * 4 + 2] = v.z;
      Ls[kk][nn + j * 4 + 3] = v.w;
    }
  }
  __syncthreads();
  {
    const int nn  = tid >> 2;
    const int kk0 = (tid & 3) * 16;
    unsigned int o[8];
#pragma unroll
    for (int j = 0; j < 8; ++j)
      o[j] = pack2(Ls[kk0 + 2 * j][nn], Ls[kk0 + 2 * j + 1][nn]);
    uint4* q = (uint4*)(Wdst + (size_t)(nBase + nn) * D_MODEL + kBase + kk0);
    q[0] = make_uint4(o[0], o[1], o[2], o[3]);
    q[1] = make_uint4(o[4], o[5], o[6], o[7]);
  }
}

// ---------------- Pass 2: fused triple GEMM + gate epilogue ----------------
extern "C" __global__ __launch_bounds__(512)
void gemm3_gate_kernel(const unsigned short* __restrict__ Xb,   // [M][K] bf16
                       const unsigned short* __restrict__ Wtb,  // 3x [N][K] bf16
                       const float* __restrict__ bf, const float* __restrict__ bi,
                       const float* __restrict__ bb,
                       float* __restrict__ Aout,     // decay a = 1 - f
                       float* __restrict__ Bgout)    // drive  i*b
{
  // Double-buffered staging: rows 0..127 = X tile, 128+w*128+n = W tiles.
  __shared__ __align__(16) unsigned short lds[2][512][LDSP];   // 80 KB

  const int tid  = threadIdx.x;
  const int lane = tid & 31;
  const int wave = tid >> 5;          // 16 waves
  const int wm   = wave & 3;          // wave M pos -> 32 rows
  const int wn   = wave >> 2;         // wave N pos -> 32 cols
  const int l15  = lane & 15;
  const int hi   = lane >> 4;

  const int mBase = blockIdx.y * BM;
  const int nBase = blockIdx.x * BN;

  // Per-thread staging row (one 64B row per thread per K-step).
  // saddr must be wave-uniform: waves 0-3 stage X, waves 4-15 stage W[(wv-4)>>2].
  const int wv = __builtin_amdgcn_readfirstlane(wave);
  const unsigned short* gbase;
  unsigned gRowOff;                       // byte offset of row start (k=0)
  if (wv < 4) {
    gbase   = Xb;
    gRowOff = (unsigned)(mBase + tid) * (D_MODEL * 2);
  } else {
    gbase   = Wtb + (size_t)((wv - 4) >> 2) * (D_MODEL * D_MODEL);
    gRowOff = (unsigned)(nBase + (tid & 127)) * (D_MODEL * 2);
  }
  const unsigned ldsRow0 = (unsigned)(uintptr_t)&lds[0][tid][0];

  v8f acc[3][2][2];
#pragma unroll
  for (int w = 0; w < 3; ++w)
#pragma unroll
    for (int sm = 0; sm < 2; ++sm)
#pragma unroll
      for (int sn = 0; sn < 2; ++sn) {
        v8f z = { 0.f, 0.f, 0.f, 0.f, 0.f, 0.f, 0.f, 0.f };
        acc[w][sm][sn] = z;
      }

  // Prologue: stage K-tile 0 into buffer 0.
  async_row64(ldsRow0, gRowOff, gbase);
  wait_async0();
  __syncthreads();

  int buf = 0;
  for (int k0 = 0; k0 < D_MODEL; k0 += BK) {
    const bool more = (k0 + BK) < D_MODEL;
    if (more)                                   // overlap next tile with WMMA
      async_row64(ldsRow0 + (unsigned)(buf ^ 1) * BUFB,
                  gRowOff + (unsigned)(k0 + BK) * 2, gbase);

    // A fragments (16-bit A 16x32: lanes0-15 K0-7/16-23, lanes16-31 K8-15/24-31)
    Frag a[2];
#pragma unroll
    for (int sm = 0; sm < 2; ++sm) {
      const unsigned short* p = &lds[buf][wm * 32 + sm * 16 + l15][hi * 8];
      a[sm].q[0] = *(const uint4*)(p);
      a[sm].q[1] = *(const uint4*)(p + 16);
    }
    // B fragments per W (lane = column; lanes0-15 K0-7/16-23, lanes16-31 K8-15/24-31)
#pragma unroll
    for (int w = 0; w < 3; ++w) {
      Frag b[2];
#pragma unroll
      for (int sn = 0; sn < 2; ++sn) {
        const unsigned short* p =
            &lds[buf][128 + w * 128 + wn * 32 + sn * 16 + l15][hi * 8];
        b[sn].q[0] = *(const uint4*)(p);
        b[sn].q[1] = *(const uint4*)(p + 16);
      }
#pragma unroll
      for (int sm = 0; sm < 2; ++sm)
#pragma unroll
        for (int sn = 0; sn < 2; ++sn)
          acc[w][sm][sn] = __builtin_amdgcn_wmma_f32_16x16x32_bf16(
              false, a[sm].v, false, b[sn].v,
              (short)0, acc[w][sm][sn], false, false);
    }

    if (more) wait_async0();    // next buffer resident before anyone reads it
    __syncthreads();            // all ds reads of 'buf' retired block-wide
    buf ^= 1;
  }

  // ---- gate epilogue straight from accumulators ----
#pragma unroll
  for (int sm = 0; sm < 2; ++sm)
#pragma unroll
    for (int sn = 0; sn < 2; ++sn) {
      const int n    = nBase + wn * 32 + sn * 16 + l15;
      const float bfv = bf[n], biv = bi[n], bbv = bb[n];
      const int mrow = mBase + wm * 32 + sm * 16 + hi * 8;
#pragma unroll
      for (int r = 0; r < 8; ++r) {
        const float yf = acc[0][sm][sn][r] + bfv;
        const float yi = acc[1][sm][sn][r] + biv;
        const float yb = (acc[2][sm][sn][r] + bbv) * 0.03125f;   // 1/sqrt(1024)
        const float av = 1.0f / (1.0f + __expf(yf));             // 1 - sigmoid(yf)
        const float ig = 1.0f / (1.0f + __expf(-yi));            // sigmoid(yi)
        const size_t idx = (size_t)(mrow + r) * D_MODEL + n;
        Aout[idx]  = av;
        Bgout[idx] = ig * yb;
      }
    }
}

// ---------------- Pass 3: h_t = a_t * h_{t-1} + b_t ----------------
extern "C" __global__ __launch_bounds__(256)
void scan_kernel(const float* __restrict__ A, float* __restrict__ HB)
{
  const int t = blockIdx.x * blockDim.x + threadIdx.x;   // 0..4095
  const int b = t >> 10;
  const int d = t & (D_MODEL - 1);
  size_t idx = (size_t)b * SEQ * D_MODEL + d;
  float h = 0.0f;
#pragma unroll 4
  for (int s = 0; s < SEQ; ++s) {
    h = A[idx] * h + HB[idx];
    HB[idx] = h;
    idx += D_MODEL;
  }
}

extern "C" void kernel_launch(void* const* d_in, const int* in_sizes, int n_in,
                              void* d_out, int out_size, void* d_ws, size_t ws_size,
                              hipStream_t stream) {
  const float* x  = (const float*)d_in[0];
  const float* Wf = (const float*)d_in[1];
  const float* bf = (const float*)d_in[2];
  const float* Wi = (const float*)d_in[3];
  const float* bi = (const float*)d_in[4];
  const float* Wb = (const float*)d_in[5];
  const float* bb = (const float*)d_in[6];
  float* out = (float*)d_out;

  // workspace layout: [0,64MB) decay A | [64,96MB) x bf16 | [96,102MB) W^T bf16
  char* ws = (char*)d_ws;
  float*          Abuf = (float*)(ws);
  unsigned short* Xb   = (unsigned short*)(ws + (size_t)64 * 1024 * 1024);
  unsigned short* Wtb  = (unsigned short*)(ws + (size_t)96 * 1024 * 1024);

  cvt_x_kernel<<<(M_TOT * D_MODEL) / (8 * 256), 256, 0, stream>>>(x, Xb);
  cvt_wt_kernel<<<dim3(256, 3), 256, 0, stream>>>(Wf, Wi, Wb, Wtb);

  dim3 grid(D_MODEL / BN, M_TOT / BM);     // (8, 128) -> 1024 workgroups
  gemm3_gate_kernel<<<grid, 512, 0, stream>>>(Xb, Wtb, bf, bi, bb, Abuf, out);

  scan_kernel<<<(BATCH * D_MODEL) / 256, 256, 0, stream>>>(Abuf, out);
}
